// SeqPairCrossUpdate_29240137351804
// MI455X (gfx1250) — matrix operand order repaired and checked by reading, not compiled, not run on hardware
//
#include <hip/hip_runtime.h>
#include <cstdint>
#include <cstddef>

// ---------------------------------------------------------------------------
// Problem constants (from reference): B=2, L=512, D=256, P=128
// ---------------------------------------------------------------------------
#define BB 2
#define LL 512
#define DD 256
#define PP 128
#define LN_EPS 1e-5f
#define JCHUNK 64   // j rows per block in the WMMA kernel

typedef __attribute__((ext_vector_type(16))) __bf16 v16bf;
typedef __attribute__((ext_vector_type(8)))  __bf16 v8bf;
typedef __attribute__((ext_vector_type(4)))  __bf16 v4bf;
typedef __attribute__((ext_vector_type(8)))  float  v8f;

__device__ __forceinline__ float wave_sum(float v) {
#pragma unroll
  for (int o = 16; o > 0; o >>= 1) v += __shfl_xor(v, o, 32);
  return v;
}

__device__ __forceinline__ float sigmoidf_dev(float x) {
  return 1.0f / (1.0f + __expf(-x));
}

__device__ __forceinline__ float geluf(float x) {
  return 0.5f * x * (1.0f + erff(x * 0.70710678f));
}

// ---------------------------------------------------------------------------
// 128-thread (4-wave) block sum reduction, broadcast to all threads
// ---------------------------------------------------------------------------
__device__ __forceinline__ float block_reduce_sum_128(float v, float* sbuf) {
  v = wave_sum(v);
  const int lane = threadIdx.x & 31;
  const int w    = threadIdx.x >> 5;
  __syncthreads();
  if (lane == 0) sbuf[w] = v;
  __syncthreads();
  return sbuf[0] + sbuf[1] + sbuf[2] + sbuf[3];
}

// ---------------------------------------------------------------------------
// Kernel 1: pair -> seq.  One block (128 thr = 4 waves) per (b,i).
// Each wave owns j = w, w+4, ... (128 rows) with an independent online
// softmax; lane owns 4 contiguous channels => all pair reads are b128.
// No block barriers in the j loop — only __shfl_xor wave reductions.
// ---------------------------------------------------------------------------
__global__ __launch_bounds__(128) void k1_pool_proj(
    const float* __restrict__ pair, const float* __restrict__ ln_p_w,
    const float* __restrict__ ln_p_b, const float* __restrict__ pool_w,
    const float* __restrict__ pool_b, const float* __restrict__ proj_w,
    const float* __restrict__ proj_b, const float* __restrict__ gate_ps,
    const float* __restrict__ seq, float* __restrict__ out_seq) {
  __shared__ float sM[4], sS[4];
  __shared__ float sAcc[4 * PP];
  __shared__ float sPool[PP];

  const int bi   = blockIdx.x;          // b*L + i
  const int lane = threadIdx.x & 31;
  const int w    = threadIdx.x >> 5;

  // lane owns channels 4*lane .. 4*lane+3
  const float4 lwv = *(const float4*)(ln_p_w + 4 * lane);
  const float4 lbv = *(const float4*)(ln_p_b + 4 * lane);
  const float4 pwv = *(const float4*)(pool_w + 4 * lane);
  const float  pb  = pool_b[0];

  const float* prow = pair + (size_t)bi * LL * PP;

  float m = -INFINITY, s = 0.0f;
  float4 acc = make_float4(0.f, 0.f, 0.f, 0.f);

  for (int j = w; j < LL; j += 4) {
    const float4 x = *(const float4*)(prow + (size_t)j * PP + 4 * lane);
    const float mu = wave_sum(x.x + x.y + x.z + x.w) * (1.0f / PP);
    const float dx = x.x - mu, dy = x.y - mu, dz = x.z - mu, dw = x.w - mu;
    const float var =
        wave_sum(dx * dx + dy * dy + dz * dz + dw * dw) * (1.0f / PP);
    const float rn = rsqrtf(var + LN_EPS);
    const float yx = dx * rn * lwv.x + lbv.x;
    const float yy = dy * rn * lwv.y + lbv.y;
    const float yz = dz * rn * lwv.z + lbv.z;
    const float yw = dw * rn * lwv.w + lbv.w;
    const float logit =
        wave_sum(yx * pwv.x + yy * pwv.y + yz * pwv.z + yw * pwv.w) + pb;
    const float mn   = fmaxf(m, logit);
    const float corr = __expf(m - mn);      // first iter: exp(-inf)=0
    const float e    = __expf(logit - mn);
    s = s * corr + e;
    acc.x = acc.x * corr + e * yx;
    acc.y = acc.y * corr + e * yy;
    acc.z = acc.z * corr + e * yz;
    acc.w = acc.w * corr + e * yw;
    m = mn;
  }

  // Merge the 4 wave-local online-softmax states.
  if (lane == 0) { sM[w] = m; sS[w] = s; }
  *(float4*)(&sAcc[w * PP + 4 * lane]) = acc;
  __syncthreads();

  const int p = threadIdx.x;              // channel 0..127
  const float M = fmaxf(fmaxf(sM[0], sM[1]), fmaxf(sM[2], sM[3]));
  float S = 0.0f, pooled = 0.0f;
#pragma unroll
  for (int w4 = 0; w4 < 4; ++w4) {
    const float f = __expf(sM[w4] - M);
    S      += sS[w4] * f;
    pooled += sAcc[w4 * PP + p] * f;
  }
  sPool[p] = pooled / S;
  __syncthreads();

  // seq_updated = seq + sigmoid(gate_ps) * (pooled @ proj_w + proj_b)
  const float sg = sigmoidf_dev(gate_ps[0]);
#pragma unroll
  for (int dcol = p; dcol < DD; dcol += PP) {
    float sum = 0.0f;
#pragma unroll 4
    for (int q = 0; q < PP; ++q) sum += sPool[q] * proj_w[q * DD + dcol];
    const size_t idx = (size_t)bi * DD + dcol;
    out_seq[idx] = seq[idx] + sg * (sum + proj_b[dcol]);
  }
}

// ---------------------------------------------------------------------------
// Kernel 2: LN(seq_updated) + two GEMVs -> hi (with b1 folded in) and hj.
// One block per (b,i); 128 threads.
// ---------------------------------------------------------------------------
__global__ __launch_bounds__(128) void k2_seqln_gemv(
    const float* __restrict__ out_seq, const float* __restrict__ ln_s_w,
    const float* __restrict__ ln_s_b, const float* __restrict__ w1,
    const float* __restrict__ b1, float* __restrict__ hi,
    float* __restrict__ hj) {
  __shared__ float sbuf[4];
  __shared__ float sx[DD];

  const int bi = blockIdx.x;
  const int t  = threadIdx.x;

  const float x0 = out_seq[(size_t)bi * DD + t];
  const float x1 = out_seq[(size_t)bi * DD + t + PP];
  const float mu = block_reduce_sum_128(x0 + x1, sbuf) * (1.0f / DD);
  const float d0 = x0 - mu, d1 = x1 - mu;
  const float var = block_reduce_sum_128(d0 * d0 + d1 * d1, sbuf) * (1.0f / DD);
  const float rn = rsqrtf(var + LN_EPS);
  sx[t]      = d0 * rn * ln_s_w[t] + ln_s_b[t];
  sx[t + PP] = d1 * rn * ln_s_w[t + PP] + ln_s_b[t + PP];
  __syncthreads();

  float ai = b1[t];
  float aj = 0.0f;
#pragma unroll 4
  for (int d = 0; d < DD; ++d) {
    const float sn = sx[d];
    ai += sn * w1[(size_t)d * PP + t];
    aj += sn * w1[(size_t)(DD + d) * PP + t];
  }
  hi[(size_t)bi * PP + t] = ai;
  hj[(size_t)bi * PP + t] = aj;
}

// ---------------------------------------------------------------------------
// Kernel 3: fused bf16-WMMA GEMM for the pair update.
//   h[j,:] = gelu(hi[b,i,:] + hj[b,j,:]);  out = pair + sg*(h @ w2 + b2)
// 256 threads = 8 waves; wave w owns N-tile w, loops 4 j-tiles x 4 K-steps
// of v_wmma_f32_16x16x32_bf16.  The 32KB w2 staging buffer is aliased as the
// f32 output tile once B-fragments live in registers, so the epilogue is one
// fully-coalesced float4 pass (ds b128 + global b128 in/out).
//
// Fragment packing used by this kernel (wave32):
//   A 16x32 bf16 : lane l holds row m=l&15; e=0..15 -> K = 16*(l>>4)+e
//                  (two 16B LDS chunks)
//   B 32x16 bf16 : lane l holds col n=l&15; K = 16*(l>>4)+e, e=0..15
//                  (one contiguous 32B chunk of the transposed-w2 LDS tile)
//   C/D 16x16 f32: reg r, lane l -> m = r + 8*(l>>4), n = l&15
// ---------------------------------------------------------------------------
__global__ __launch_bounds__(256) void k3_pair_update(
    const float* __restrict__ pair, const float* __restrict__ hi,
    const float* __restrict__ hj, const float* __restrict__ w2,
    const float* __restrict__ b2, const float* __restrict__ gate_sp,
    float* __restrict__ out_pair) {
  __shared__ union __align__(32) {
    __bf16 w[PP * PP];            // phase 1: transposed bf16 w2 (32 KB)
    float  out[JCHUNK * PP];      // phase 3: f32 output tile    (32 KB)
  } sU;
  __shared__ __align__(32) __bf16 sA[JCHUNK * PP];  // h tile [j_local][k]
  __shared__ float sHi[PP];
  __shared__ float sB2[PP];

  const int t   = threadIdx.x;
  const int bid = blockIdx.x;
  const int jc  = bid & 7;              // 8 chunks of 64 rows
  const int bi  = bid >> 3;             // b*L + i
  const int b   = bi >> 9;              // / L
  const int j0  = jc * JCHUNK;

  // ---- Phase 1: stage transposed bf16 w2 (b128 global reads) -------------
  for (int e4 = t * 4; e4 < PP * PP; e4 += 1024) {
    const int k = e4 >> 7, n = e4 & (PP - 1);
    const float4 wv = *(const float4*)(w2 + e4);
    sU.w[(n + 0) * PP + k] = (__bf16)wv.x;
    sU.w[(n + 1) * PP + k] = (__bf16)wv.y;
    sU.w[(n + 2) * PP + k] = (__bf16)wv.z;
    sU.w[(n + 3) * PP + k] = (__bf16)wv.w;
  }
  if (t < PP) {
    sHi[t] = hi[(size_t)bi * PP + t];   // b1 already folded in
    sB2[t] = b2[t];
  }
  __syncthreads();

  const int lane = t & 31;
  const int half = lane >> 4;           // 0: lanes 0-15, 1: lanes 16-31
  const int lw   = lane & 15;
  const int nt   = t >> 5;              // wave id == N-tile id (0..7)

  // ---- Phase 2a: preload B fragments (reused by every j-tile) ------------
  v16bf bfrag[4];
  {
    const __bf16* wrow = &sU.w[(nt * 16 + lw) * PP];
#pragma unroll
    for (int ks = 0; ks < 4; ++ks) {
      bfrag[ks] = *(const v16bf*)(wrow + ks * 32 + half * 16);  // 32B aligned
    }
  }

  // ---- Phase 2b: gelu A-tile, bf16 into LDS (b128 global reads) ----------
  for (int e4 = t * 4; e4 < JCHUNK * PP; e4 += 1024) {
    const int jl = e4 >> 7, k = e4 & (PP - 1);
    const float4 hv =
        *(const float4*)(hj + ((size_t)(b * LL + j0 + jl)) * PP + k);
    v4bf g;
    g[0] = (__bf16)geluf(sHi[k + 0] + hv.x);
    g[1] = (__bf16)geluf(sHi[k + 1] + hv.y);
    g[2] = (__bf16)geluf(sHi[k + 2] + hv.z);
    g[3] = (__bf16)geluf(sHi[k + 3] + hv.w);
    *(v4bf*)(&sA[e4]) = g;
  }
  __syncthreads();   // sA ready; everyone done reading sU.w

  // ---- Phase 3: WMMA loops; acc (+bias) staged into aliased f32 tile -----
  const int col = nt * 16 + lw;
  const float bias = sB2[col];
#pragma unroll
  for (int jt = 0; jt < 4; ++jt) {
    const __bf16* arow = &sA[(jt * 16 + lw) * PP];
    // Load all A fragments first so the 4 wmmas can issue back-to-back.
    v16bf afrag[4];
#pragma unroll
    for (int ks = 0; ks < 4; ++ks) {
      const v8bf a0 = *(const v8bf*)(arow + ks * 32 + half * 16);      // K 16h+e, e=0..7
      const v8bf a1 = *(const v8bf*)(arow + ks * 32 + half * 16 + 8);  // K 16h+e, e=8..15
      afrag[ks] = __builtin_shufflevector(
          a0, a1, 0, 1, 2, 3, 4, 5, 6, 7, 8, 9, 10, 11, 12, 13, 14, 15);
    }
    v8f acc = {};
#pragma unroll
    for (int ks = 0; ks < 4; ++ks) {
      acc = __builtin_amdgcn_wmma_f32_16x16x32_bf16(
          /*neg_a=*/false, afrag[ks], /*neg_b=*/false, bfrag[ks],
          /*c_mod=*/(short)0, acc, /*reuse_a=*/false, /*reuse_b=*/false);
    }
    // m = r + 8*half, n = lw within the tile; fold bias here.
#pragma unroll
    for (int r = 0; r < 8; ++r) {
      sU.out[(jt * 16 + r + 8 * half) * PP + col] = acc[r] + bias;
    }
  }
  __syncthreads();

  // ---- Phase 4: coalesced gated-residual epilogue (all b128) -------------
  const float sg = sigmoidf_dev(gate_sp[0]);
  for (int e4 = t * 4; e4 < JCHUNK * PP; e4 += 1024) {
    const int jl = e4 >> 7, k = e4 & (PP - 1);
    const size_t idx = ((size_t)bi * LL + j0 + jl) * PP + k;
    const float4 rv = *(const float4*)(pair + idx);
    const float4 ov = *(const float4*)(&sU.out[e4]);
    float4 o;
    o.x = rv.x + sg * ov.x;
    o.y = rv.y + sg * ov.y;
    o.z = rv.z + sg * ov.z;
    o.w = rv.w + sg * ov.w;
    *(float4*)(out_pair + idx) = o;
  }
}

// ---------------------------------------------------------------------------
// Host launcher
// ---------------------------------------------------------------------------
extern "C" void kernel_launch(void* const* d_in, const int* in_sizes, int n_in,
                              void* d_out, int out_size, void* d_ws,
                              size_t ws_size, hipStream_t stream) {
  const float* seq     = (const float*)d_in[0];
  const float* pair    = (const float*)d_in[1];
  const float* ln_p_w  = (const float*)d_in[2];
  const float* ln_p_b  = (const float*)d_in[3];
  const float* pool_w  = (const float*)d_in[4];
  const float* pool_b  = (const float*)d_in[5];
  const float* proj_w  = (const float*)d_in[6];
  const float* proj_b  = (const float*)d_in[7];
  const float* gate_ps = (const float*)d_in[8];
  const float* ln_s_w  = (const float*)d_in[9];
  const float* ln_s_b  = (const float*)d_in[10];
  const float* w1      = (const float*)d_in[11];
  const float* b1      = (const float*)d_in[12];
  const float* w2      = (const float*)d_in[13];
  const float* b2      = (const float*)d_in[14];
  const float* gate_sp = (const float*)d_in[15];

  float* out_seq  = (float*)d_out;                       // (B,L,D)
  float* out_pair = out_seq + (size_t)BB * LL * DD;      // (B,L,L,P)

  float* hi = (float*)d_ws;                              // (B,L,P)
  float* hj = hi + (size_t)BB * LL * PP;                 // (B,L,P)

  k1_pool_proj<<<BB * LL, 128, 0, stream>>>(pair, ln_p_w, ln_p_b, pool_w,
                                            pool_b, proj_w, proj_b, gate_ps,
                                            seq, out_seq);
  k2_seqln_gemv<<<BB * LL, 128, 0, stream>>>(out_seq, ln_s_w, ln_s_b, w1, b1,
                                             hi, hj);
  k3_pair_update<<<BB * LL * (LL / JCHUNK), 256, 0, stream>>>(
      pair, hi, hj, w2, b2, gate_sp, out_pair);
}
